// SeasonalFrequencyMLP_8718783611600
// MI455X (gfx1250) — matrix operand order repaired and checked by reading, not compiled, not run on hardware
//
#include <hip/hip_runtime.h>

// ---------------------------------------------------------------------------
// SeasonalFrequencyMLP on MI455X (gfx1250, wave32, WMMA f16 16x16x32 / f32 acc)
// FFT expressed as dense DFT GEMMs so every stage runs on the matrix pipes.
// Frequency-domain complex GEMM streams its 134 MB of weights through LDS via
// the Tensor Data Mover (TENSOR_LOAD_TO_LDS, TENSORcnt), double-buffered.
// ---------------------------------------------------------------------------

typedef _Float16 half_t;
typedef __attribute__((ext_vector_type(16))) _Float16 v16h;
typedef __attribute__((ext_vector_type(8)))  _Float16 v8h;
typedef __attribute__((ext_vector_type(8)))  float    v8f;
typedef __attribute__((ext_vector_type(4)))  unsigned int u32x4;
typedef __attribute__((ext_vector_type(8)))  int i32x8;
typedef __attribute__((ext_vector_type(4)))  int i32x4;

#define B_SZ  32
#define SEQ   1024
#define D_IN  128
#define D_HID 256
#define D_OUT 128
#define NF    513          // SEQ/2 + 1
#define FP    544          // NF padded to multiple of 32 (17 K-chunks)

// TDM chunk: 32 K-rows x 256 cols of f32 = 32 KB
#define TDM_ROWS   32
#define TDM_CHUNK  (TDM_ROWS * D_HID * 4)

// ---------------- WMMA wrapper (codegen-confirmed builtin) -----------------
__device__ __forceinline__ v8f wmma16(v16h a, v16h b, v8f c) {
  // D = A(16x32 f16) * B(32x16 f16) + C(16x16 f32)
  return __builtin_amdgcn_wmma_f32_16x16x32_f16(
      /*neg_a=*/false, a, /*neg_b=*/false, b,
      /*c_mod=*/(short)0, c, /*reuse_a=*/false, /*reuse_b=*/false);
}

__device__ __forceinline__ v8f vzero8() {
  v8f z;
#pragma unroll
  for (int i = 0; i < 8; ++i) z[i] = 0.0f;
  return z;
}

__device__ __forceinline__ v16h vneg16(v16h a) {
  v16h r;
#pragma unroll
  for (int i = 0; i < 16; ++i) r[i] = -a[i];
  return r;
}

// ---- A fragment (16x32, row-major source, origin pre-offset to row0/k0) ----
// Lane L: m = L&15, hi = L>>4.  VGPR j<4 : K = 8*hi + 2j{,+1}
//                               VGPR j>=4: K = 16 + 8*hi + 2(j-4){,+1}
__device__ __forceinline__ v16h load_A_frag(const half_t* A, int lda, int lane) {
  const int m = lane & 15, hi = lane >> 4;
  const half_t* p = A + (size_t)m * lda + 8 * hi;
  union { v16h v; v8h h[2]; } u;
  u.h[0] = *(const v8h*)(p);
  u.h[1] = *(const v8h*)(p + 16);
  return u.v;
}

// ---- B fragment (32x16, row-major [K][N] source, origin at k0 row/n0 col) --
__device__ __forceinline__ v16h load_B_frag_h(const half_t* B, int ldb, int lane) {
  const int n = lane & 15, hi = lane >> 4;
  const half_t* p = B + n;
  v16h r;
#pragma unroll
  for (int s = 0; s < 8; ++s) {
    r[s]     = p[(size_t)(8 * hi + s) * ldb];
    r[8 + s] = p[(size_t)(16 + 8 * hi + s) * ldb];
  }
  return r;
}

// ---- B fragment from f32 source (global or LDS): strided f32 loads + cvt to f16 ---
__device__ __forceinline__ v16h load_B_frag_f32(const float* B, int ldb, int lane) {
  const int n = lane & 15, hi = lane >> 4;
  const float* p = B + n;
  v16h r;
#pragma unroll
  for (int s = 0; s < 8; ++s) {
    r[s]     = (_Float16)p[(size_t)(8 * hi + s) * ldb];
    r[8 + s] = (_Float16)p[(size_t)(16 + 8 * hi + s) * ldb];
  }
  return r;
}

// ---------------------------------------------------------------------------
// TDM: DMA a 2-D tile [rows x 256 f32] from global into LDS at byte offset
// lds_off.  D# per CDNA5 ISA (group0 + group1, 2-D form,
// groups 2/3/4 zero).  Issued once per wave (EXEC ignored); tracked by
// TENSORcnt.  6-arg builtin form (clang-23 / therock headers).
// ---------------------------------------------------------------------------
__device__ __forceinline__ void tdm_load_2d_f32(const float* gsrc, unsigned lds_off,
                                                unsigned rows) {
  const unsigned long long ga = (unsigned long long)(const void*)gsrc;
  u32x4 g0;
  g0[0] = 1u;                                         // count=1, user descriptor
  g0[1] = lds_off;                                    // lds_addr (bytes)
  g0[2] = (unsigned)(ga & 0xFFFFFFFFu);               // global_addr[31:0]
  g0[3] = (unsigned)((ga >> 32) & 0x1FFFFFFu)         // global_addr[56:32]
        | (2u << 30);                                 // type = 2 ("image")
  i32x8 g1;
  g1[0] = (int)(2u << 16);                            // wg_mask=0, data_size=2 (4B)
  g1[1] = (int)((unsigned)D_HID << 16);               // tensor_dim0[15:0] in [31:16]
  g1[2] = (int)((unsigned)rows << 16);                // dim0 hi=0 | tensor_dim1 lo16
  g1[3] = (int)((unsigned)D_HID << 16);               // dim1 hi=0 | tile_dim0=256
  g1[4] = (int)(rows & 0xFFFFu);                      // tile_dim1=rows, tile_dim2=0
  g1[5] = (int)D_HID;                                 // tensor_dim0_stride lo32
  g1[6] = 0;                                          // stride hi | dim1_stride lo
  g1[7] = 0;
  i32x4 z4;  z4[0] = 0; z4[1] = 0; z4[2] = 0; z4[3] = 0;   // groups 2/3 unused
  i32x8 z8;
#pragma unroll
  for (int i = 0; i < 8; ++i) z8[i] = 0;                   // extra group unused
  __builtin_amdgcn_tensor_load_to_lds(g0, g1, z4, z4, z8, 0);
}

// ---------------------------------------------------------------------------
// K1: build DFT/iDFT trig tables (f16) and zero the padded Y buffers.
// ---------------------------------------------------------------------------
__global__ void init_tables(half_t* __restrict__ A1c, half_t* __restrict__ A1s,
                            half_t* __restrict__ Cc,  half_t* __restrict__ Cs,
                            half_t* __restrict__ Yre, half_t* __restrict__ Yim) {
  const float TWO_PI = 6.28318530717958647692f;
  const int stride = gridDim.x * blockDim.x;
  const int tid0 = blockIdx.x * blockDim.x + threadIdx.x;

  for (int idx = tid0; idx < FP * SEQ; idx += stride) {
    int f = idx / SEQ, t = idx % SEQ;
    float c = 0.0f, s = 0.0f;
    if (f < NF) {
      int r = (f * t) & (SEQ - 1);
      float ang = TWO_PI * (float)r / (float)SEQ;
      c = __cosf(ang);
      s = -__sinf(ang);
    }
    A1c[idx] = (half_t)c;
    A1s[idx] = (half_t)s;
  }

  for (int idx = tid0; idx < SEQ * FP; idx += stride) {
    int t = idx / FP, f = idx % FP;
    float cc = 0.0f, ss = 0.0f;
    if (f == 0) {
      cc = 1.0f / (float)SEQ;
    } else if (f < SEQ / 2) {
      int r = (f * t) & (SEQ - 1);
      float ang = TWO_PI * (float)r / (float)SEQ;
      cc =  2.0f * __cosf(ang) / (float)SEQ;
      ss = -2.0f * __sinf(ang) / (float)SEQ;
    } else if (f == SEQ / 2) {
      cc = ((t & 1) ? -1.0f : 1.0f) / (float)SEQ;
    }
    Cc[idx] = (half_t)cc;
    Cs[idx] = (half_t)ss;
  }

  for (int idx = tid0; idx < B_SZ * FP * D_HID; idx += stride) {
    Yre[idx] = (half_t)0.0f;
    Yim[idx] = (half_t)0.0f;
  }
}

// ---------------------------------------------------------------------------
// K2: cast x (f32) -> xh (f16), same [b][t][i] layout.
// ---------------------------------------------------------------------------
__global__ void cast_x(const float* __restrict__ x, half_t* __restrict__ xh, int n) {
  const int stride = gridDim.x * blockDim.x;
  for (int i = blockIdx.x * blockDim.x + threadIdx.x; i < n; i += stride)
    xh[i] = (half_t)x[i];
}

// ---------------------------------------------------------------------------
// K3: forward DFT as GEMM.  Per batch b: M=FP(freq), K=SEQ(t), N=D_IN(i).
// ---------------------------------------------------------------------------
__global__ void dft_gemm(const half_t* __restrict__ A1c, const half_t* __restrict__ A1s,
                         const half_t* __restrict__ xh,
                         half_t* __restrict__ Xre, half_t* __restrict__ Xim) {
  const int lane = threadIdx.x & 31;
  const int wave = blockIdx.x * (blockDim.x >> 5) + (threadIdx.x >> 5);
  const int FT = FP / 16;                 // 34
  const int b  = wave / (FT * 2);
  const int rr = wave % (FT * 2);
  const int ft = rr >> 1;
  const int ih = rr & 1;                  // which 64-wide half of D_IN

  v8f acc_re[4], acc_im[4];
#pragma unroll
  for (int n = 0; n < 4; ++n) { acc_re[n] = vzero8(); acc_im[n] = vzero8(); }

  for (int kc = 0; kc < SEQ / 32; ++kc) {
    const int k0 = kc * 32;
    v16h ac = load_A_frag(A1c + (size_t)ft * 16 * SEQ + k0, SEQ, lane);
    v16h as = load_A_frag(A1s + (size_t)ft * 16 * SEQ + k0, SEQ, lane);
#pragma unroll
    for (int n = 0; n < 4; ++n) {
      const half_t* bp = xh + ((size_t)b * SEQ + k0) * D_IN + ih * 64 + n * 16;
      v16h bb = load_B_frag_h(bp, D_IN, lane);
      acc_re[n] = wmma16(ac, bb, acc_re[n]);
      acc_im[n] = wmma16(as, bb, acc_im[n]);
    }
  }

  const int hi = lane >> 4, nl = lane & 15;
#pragma unroll
  for (int n = 0; n < 4; ++n) {
    const int col = ih * 64 + n * 16 + nl;
#pragma unroll
    for (int r8 = 0; r8 < 8; ++r8) {
      const int f = ft * 16 + hi * 8 + r8;
      Xre[((size_t)b * FP + f) * D_IN + col] = (half_t)acc_re[n][r8];
      Xim[((size_t)b * FP + f) * D_IN + col] = (half_t)acc_im[n][r8];
    }
  }
}

// ---------------------------------------------------------------------------
// K4: per-frequency complex GEMM + bias + split-ReLU, TDM-fed.
// One block (8 waves) per frequency f: M=32(b), K=128(i), N=256(o).
// W_re/W_im K-chunks are DMA'd into double-buffered LDS by the Tensor Data
// Mover (wave 0 issues, s_wait_tensorcnt + barrier hands off to all 8 waves),
// so the 134 MB weight stream never occupies the WMMA waves' vector-load slots.
// ---------------------------------------------------------------------------
__global__ void freq_cgemm(const half_t* __restrict__ Xre, const half_t* __restrict__ Xim,
                           const float* __restrict__ W_re, const float* __restrict__ W_im,
                           const float* __restrict__ B_re, const float* __restrict__ B_im,
                           half_t* __restrict__ Yre, half_t* __restrict__ Yim) {
  extern __shared__ __align__(16) char smem[];   // 2 buffers x (Re 32KB + Im 32KB)

  const int f    = blockIdx.x;            // 0..512
  const int lane = threadIdx.x & 31;
  const int nt   = threadIdx.x >> 5;      // 0..7, 32 output cols each
  const int n0   = nt * 32;
  const int ldA  = FP * D_IN;             // batch stride of X arrays
  const bool issuer = (threadIdx.x < 32); // wave 0 drives the TDM

  const float* Wre_f = W_re + (size_t)f * D_IN * D_HID;
  const float* Wim_f = W_im + (size_t)f * D_IN * D_HID;

  // Dynamic LDS is assumed to start at LDS offset 0 (no static LDS in kernel).
  auto lds_re_off = [&](int par) -> unsigned { return (unsigned)(par * 2 * TDM_CHUNK); };
  auto lds_im_off = [&](int par) -> unsigned { return (unsigned)(par * 2 * TDM_CHUNK + TDM_CHUNK); };
  auto buf_re = [&](int par) -> const float* { return (const float*)(smem + par * 2 * TDM_CHUNK); };
  auto buf_im = [&](int par) -> const float* { return (const float*)(smem + par * 2 * TDM_CHUNK + TDM_CHUNK); };

  // Prologue: DMA chunk 0 into buffer 0.
  if (issuer) {
    tdm_load_2d_f32(Wre_f, lds_re_off(0), TDM_ROWS);
    tdm_load_2d_f32(Wim_f, lds_im_off(0), TDM_ROWS);
  }

  v8f aRe[2][2], aIm[2][2];
#pragma unroll
  for (int mt = 0; mt < 2; ++mt)
#pragma unroll
    for (int nn = 0; nn < 2; ++nn) { aRe[mt][nn] = vzero8(); aIm[mt][nn] = vzero8(); }

  for (int kc = 0; kc < D_IN / TDM_ROWS; ++kc) {     // 4 chunks
    const int k0 = kc * TDM_ROWS;
    const int par = kc & 1;

    if (issuer) __builtin_amdgcn_s_wait_tensorcnt(0); // chunk kc landed in LDS
    __syncthreads();                                  // publish to all 8 waves

    // Overlap: DMA chunk kc+1 into the other buffer while computing chunk kc.
    if (issuer && kc + 1 < D_IN / TDM_ROWS) {
      const size_t goff = (size_t)(k0 + TDM_ROWS) * D_HID;
      tdm_load_2d_f32(Wre_f + goff, lds_re_off(par ^ 1), TDM_ROWS);
      tdm_load_2d_f32(Wim_f + goff, lds_im_off(par ^ 1), TDM_ROWS);
    }

    v16h axr[2], axi[2], axin[2];
#pragma unroll
    for (int mt = 0; mt < 2; ++mt) {
      const size_t org = (size_t)(mt * 16) * ldA + (size_t)f * D_IN + k0;
      axr[mt]  = load_A_frag(Xre + org, ldA, lane);
      axi[mt]  = load_A_frag(Xim + org, ldA, lane);
      axin[mt] = vneg16(axi[mt]);
    }
#pragma unroll
    for (int nn = 0; nn < 2; ++nn) {
      v16h br = load_B_frag_f32(buf_re(par) + n0 + nn * 16, D_HID, lane);
      v16h bi = load_B_frag_f32(buf_im(par) + n0 + nn * 16, D_HID, lane);
#pragma unroll
      for (int mt = 0; mt < 2; ++mt) {
        aRe[mt][nn] = wmma16(axr[mt],  br, aRe[mt][nn]);
        aRe[mt][nn] = wmma16(axin[mt], bi, aRe[mt][nn]);   // - Xim*Wim
        aIm[mt][nn] = wmma16(axr[mt],  bi, aIm[mt][nn]);
        aIm[mt][nn] = wmma16(axi[mt],  br, aIm[mt][nn]);
      }
    }
    __syncthreads();   // everyone done reading buf(par) before it is re-filled
  }

  const int hi = lane >> 4, nl = lane & 15;
#pragma unroll
  for (int nn = 0; nn < 2; ++nn) {
    const int col = n0 + nn * 16 + nl;
    const float biasR = B_re[(size_t)f * D_HID + col];
    const float biasI = B_im[(size_t)f * D_HID + col];
#pragma unroll
    for (int mt = 0; mt < 2; ++mt) {
#pragma unroll
      for (int r8 = 0; r8 < 8; ++r8) {
        const int bb = mt * 16 + hi * 8 + r8;
        const float vr = fmaxf(aRe[mt][nn][r8] + biasR, 0.0f);
        const float vi = fmaxf(aIm[mt][nn][r8] + biasI, 0.0f);
        Yre[((size_t)bb * FP + f) * D_HID + col] = (half_t)vr;
        Yim[((size_t)bb * FP + f) * D_HID + col] = (half_t)vi;
      }
    }
  }
}

// ---------------------------------------------------------------------------
// K5: inverse DFT GEMM fused with residual projection x@W_rc.
// ---------------------------------------------------------------------------
__global__ void idft_gemm(const half_t* __restrict__ Cc,  const half_t* __restrict__ Cs,
                          const half_t* __restrict__ Yre, const half_t* __restrict__ Yim,
                          const half_t* __restrict__ xh,  const float* __restrict__ W_rc,
                          half_t* __restrict__ Sh) {
  const int lane = threadIdx.x & 31;
  const int wave = blockIdx.x * (blockDim.x >> 5) + (threadIdx.x >> 5);
  const int nq = wave & 3;            // 64-wide N quad
  const int tT = (wave >> 2) & 63;    // 16-row t tile
  const int b  = wave >> 8;
  const int n0 = nq * 64;

  v8f acc[4];
#pragma unroll
  for (int n = 0; n < 4; ++n) acc[n] = vzero8();

  for (int kc = 0; kc < FP / 32; ++kc) {
    const int k0 = kc * 32;
    v16h ac = load_A_frag(Cc + (size_t)tT * 16 * FP + k0, FP, lane);
    v16h as = load_A_frag(Cs + (size_t)tT * 16 * FP + k0, FP, lane);
#pragma unroll
    for (int n = 0; n < 4; ++n) {
      v16h br = load_B_frag_h(Yre + ((size_t)b * FP + k0) * D_HID + n0 + n * 16, D_HID, lane);
      v16h bi = load_B_frag_h(Yim + ((size_t)b * FP + k0) * D_HID + n0 + n * 16, D_HID, lane);
      acc[n] = wmma16(ac, br, acc[n]);
      acc[n] = wmma16(as, bi, acc[n]);
    }
  }

  for (int kc = 0; kc < D_IN / 32; ++kc) {
    const int k0 = kc * 32;
    v16h ax = load_A_frag(xh + ((size_t)b * SEQ + tT * 16) * D_IN + k0, D_IN, lane);
#pragma unroll
    for (int n = 0; n < 4; ++n) {
      v16h bw = load_B_frag_f32(W_rc + (size_t)k0 * D_HID + n0 + n * 16, D_HID, lane);
      acc[n] = wmma16(ax, bw, acc[n]);
    }
  }

  const int hi = lane >> 4, nl = lane & 15;
#pragma unroll
  for (int n = 0; n < 4; ++n) {
    const int col = n0 + n * 16 + nl;
#pragma unroll
    for (int r8 = 0; r8 < 8; ++r8) {
      const int t = tT * 16 + hi * 8 + r8;
      Sh[((size_t)b * SEQ + t) * D_HID + col] = (half_t)acc[n][r8];
    }
  }
}

// ---------------------------------------------------------------------------
// K6: fused MLP head. Block = (b, 16-row t tile), 8 waves.
// ---------------------------------------------------------------------------
__global__ void mlp_fused(const half_t* __restrict__ Sh,
                          const float* __restrict__ fc1_w, const float* __restrict__ fc1_b,
                          const float* __restrict__ fc2_w, const float* __restrict__ fc2_b,
                          float* __restrict__ out) {
  __shared__ __align__(16) half_t hsh[16 * D_HID];

  const int lane = threadIdx.x & 31;
  const int nt   = threadIdx.x >> 5;   // 0..7
  const int b    = blockIdx.x >> 6;
  const int tT   = blockIdx.x & 63;
  const int hi = lane >> 4, nl = lane & 15;

  v8f acc1[2];
#pragma unroll
  for (int nn = 0; nn < 2; ++nn) acc1[nn] = vzero8();

  for (int kc = 0; kc < D_HID / 32; ++kc) {
    const int k0 = kc * 32;
    v16h a = load_A_frag(Sh + ((size_t)b * SEQ + tT * 16) * D_HID + k0, D_HID, lane);
#pragma unroll
    for (int nn = 0; nn < 2; ++nn) {
      v16h bw = load_B_frag_f32(fc1_w + (size_t)k0 * D_HID + nt * 32 + nn * 16, D_HID, lane);
      acc1[nn] = wmma16(a, bw, acc1[nn]);
    }
  }

#pragma unroll
  for (int nn = 0; nn < 2; ++nn) {
    const int col = nt * 32 + nn * 16 + nl;
    const float bb = fc1_b[col];
#pragma unroll
    for (int r8 = 0; r8 < 8; ++r8) {
      float v = acc1[nn][r8] + bb;
      v = (v > 0.0f) ? v : 0.01f * v;          // LeakyReLU(0.01)
      hsh[(size_t)(hi * 8 + r8) * D_HID + col] = (half_t)v;
    }
  }
  __syncthreads();

  v8f acc2 = vzero8();
  for (int kc = 0; kc < D_HID / 32; ++kc) {
    const int k0 = kc * 32;
    v16h a  = load_A_frag(hsh + k0, D_HID, lane);
    v16h bw = load_B_frag_f32(fc2_w + (size_t)k0 * D_OUT + nt * 16, D_OUT, lane);
    acc2 = wmma16(a, bw, acc2);
  }

  {
    const int col = nt * 16 + nl;
    const float bb = fc2_b[col];
#pragma unroll
    for (int r8 = 0; r8 < 8; ++r8) {
      const int t = tT * 16 + hi * 8 + r8;
      out[((size_t)b * SEQ + t) * D_OUT + col] = acc2[r8] + bb;
    }
  }
}

// ---------------------------------------------------------------------------
extern "C" void kernel_launch(void* const* d_in, const int* in_sizes, int n_in,
                              void* d_out, int out_size, void* d_ws, size_t ws_size,
                              hipStream_t stream) {
  const float* x     = (const float*)d_in[0];
  const float* W_re  = (const float*)d_in[1];
  const float* W_im  = (const float*)d_in[2];
  const float* B_re  = (const float*)d_in[3];
  const float* B_im  = (const float*)d_in[4];
  const float* W_rc  = (const float*)d_in[5];
  const float* fc1_w = (const float*)d_in[6];
  const float* fc1_b = (const float*)d_in[7];
  const float* fc2_w = (const float*)d_in[8];
  const float* fc2_b = (const float*)d_in[9];
  float* out = (float*)d_out;

  // ---- workspace carve-up (256B aligned, ~56 MB total) ----
  char* ws = (char*)d_ws;
  size_t off = 0;
  auto carve = [&](size_t bytes) -> void* {
    void* p = ws + off;
    off += (bytes + 255) & ~(size_t)255;
    return p;
  };
  half_t* A1c = (half_t*)carve((size_t)FP * SEQ * sizeof(half_t));
  half_t* A1s = (half_t*)carve((size_t)FP * SEQ * sizeof(half_t));
  half_t* Cc  = (half_t*)carve((size_t)SEQ * FP * sizeof(half_t));
  half_t* Cs  = (half_t*)carve((size_t)SEQ * FP * sizeof(half_t));
  half_t* xh  = (half_t*)carve((size_t)B_SZ * SEQ * D_IN * sizeof(half_t));
  half_t* Xre = (half_t*)carve((size_t)B_SZ * FP * D_IN * sizeof(half_t));
  half_t* Xim = (half_t*)carve((size_t)B_SZ * FP * D_IN * sizeof(half_t));
  half_t* Yre = (half_t*)carve((size_t)B_SZ * FP * D_HID * sizeof(half_t));
  half_t* Yim = (half_t*)carve((size_t)B_SZ * FP * D_HID * sizeof(half_t));
  half_t* Sh  = (half_t*)carve((size_t)B_SZ * SEQ * D_HID * sizeof(half_t));

  // K1: trig tables + zero padded Y
  init_tables<<<512, 256, 0, stream>>>(A1c, A1s, Cc, Cs, Yre, Yim);

  // K2: x -> f16
  cast_x<<<1024, 256, 0, stream>>>(x, xh, B_SZ * SEQ * D_IN);

  // K3: forward DFT GEMM.  waves = 32 b * 34 ftiles * 2 ihalves = 2176 -> 272 blocks
  dft_gemm<<<(B_SZ * (FP / 16) * 2) / 8, 256, 0, stream>>>(A1c, A1s, xh, Xre, Xim);

  // K4: per-frequency complex GEMM, TDM-fed, 128 KB dynamic LDS (2 x (Re+Im))
  freq_cgemm<<<NF, 256, 4 * TDM_CHUNK, stream>>>(Xre, Xim, W_re, W_im,
                                                 B_re, B_im, Yre, Yim);

  // K5: iDFT GEMM + residual.  waves = 32 b * 64 ttiles * 4 nquads = 8192 -> 1024 blocks
  idft_gemm<<<1024, 256, 0, stream>>>(Cc, Cs, Yre, Yim, xh, W_rc, Sh);

  // K6: fused MLP head. blocks = 32 b * 64 ttiles = 2048
  mlp_fused<<<B_SZ * 64, 256, 0, stream>>>(Sh, fc1_w, fc1_b, fc2_w, fc2_b, out);
}
